// InvertibleResBlockBase_20950850469949
// MI455X (gfx1250) — matrix operand order repaired and checked by reading, not compiled, not run on hardware
//
#include <hip/hip_runtime.h>
#include <hip/hip_bf16.h>
#include <math.h>

typedef __bf16 bf16_t;
typedef unsigned int u32;
typedef __attribute__((ext_vector_type(16))) __bf16 v16bf;
typedef __attribute__((ext_vector_type(8)))  __bf16 v8bf;
typedef __attribute__((ext_vector_type(8)))  float  v8f;
typedef __attribute__((ext_vector_type(4)))  u32    v4u;
typedef __attribute__((ext_vector_type(8)))  int    v8i;
typedef __attribute__((ext_vector_type(4)))  int    v4i;

#ifndef __has_builtin
#define __has_builtin(x) 0
#endif
#if __has_builtin(__builtin_amdgcn_tensor_load_to_lds) && __has_builtin(__builtin_amdgcn_s_wait_tensorcnt)
#define HAVE_TDM 1
#else
#define HAVE_TDM 0
#endif
#if defined(__has_include)
#if __has_include(<hip/amd_detail/amd_gfx1250_TDM.h>)
#define TDM_6ARG 1
#else
#define TDM_6ARG 0
#endif
#else
#define TDM_6ARG 0
#endif

#define LDSTRIDE 40   // 32 data elems + 4-DWORD pad: 80B rows, 16B aligned, bank-spread
#define BK 32

// ---------------------------------------------------------------------------
// Small helper kernels
// ---------------------------------------------------------------------------
__global__ void k_zero_f32(float* __restrict__ p, int n) {
  int i = blockIdx.x * blockDim.x + threadIdx.x;
  if (i < n) p[i] = 0.0f;
}

__global__ void k_cvt_bf16(const float* __restrict__ in, bf16_t* __restrict__ out, long n) {
  long i = (long)blockIdx.x * blockDim.x + threadIdx.x;
  if (i < n) out[i] = (bf16_t)in[i];
}

// in: [R,C] f32 row-major  ->  out: [C,R] bf16 row-major
__global__ void k_cvt_tr_bf16(const float* __restrict__ in, bf16_t* __restrict__ out,
                              int R, int C) {
  long idx = (long)blockIdx.x * blockDim.x + threadIdx.x;
  if (idx < (long)R * C) {
    int i = (int)(idx / C);
    int j = (int)(idx % C);
    out[(long)j * R + i] = (bf16_t)in[idx];
  }
}

__global__ void k_final(const float* __restrict__ logdf, const float* __restrict__ sd,
                        float* __restrict__ out, int n) {
  int i = blockIdx.x * blockDim.x + threadIdx.x;
  if (i < n) out[i] = logdf[i] + sd[i];
}

// ---------------------------------------------------------------------------
// Tensor Data Mover: 2D tile (tile_d0 x tile_d1, 2-byte elems) -> padded LDS
// ---------------------------------------------------------------------------
#if HAVE_TDM
__device__ __forceinline__ u32 lds_byte_off(const void* p) {
  // generic LDS address: low 32 bits are the LDS byte offset (ISA 10.2)
  return (u32)(unsigned long long)p;
}

__device__ __forceinline__ void tdm_load_2d(u32 lds_off, const void* gaddr,
                                            u32 tile_d0, u32 tile_d1,
                                            u32 tensor_d0, u32 tensor_d1,
                                            u32 stride_d0) {
  unsigned long long ga = (unsigned long long)gaddr;
  v4u g0;
  g0[0] = 1u;                                    // count=1, user descriptor
  g0[1] = lds_off;                               // lds_addr (bytes)
  g0[2] = (u32)(ga & 0xffffffffull);             // global_addr[31:0]
  g0[3] = (u32)((ga >> 32) & 0x1ffffffull)       // global_addr[56:32]
        | (2u << 30);                            // type = 2 ("image")
  v8i g1;
  g1[0] = (int)((1u << 16)                       // data_size = 1 -> 2 bytes
              | (1u << 20)                       // pad_enable
              | (3u << 22)                       // pad_interval: 16 DWORDs (one 32-elem row)
              | (3u << 25));                     // pad_amount:   4 DWORDs (8 elems)
  g1[1] = (int)((tensor_d0 & 0xffffu) << 16);    // abar addr=0 | tensor_dim0[15:0]
  g1[2] = (int)(((tensor_d0 >> 16) & 0xffffu) | ((tensor_d1 & 0xffffu) << 16));
  g1[3] = (int)(((tensor_d1 >> 16) & 0xffffu) | ((tile_d0 & 0xffffu) << 16));
  g1[4] = (int)(tile_d1 & 0xffffu);              // tile_dim1 | tile_dim2=0
  g1[5] = (int)stride_d0;                        // tensor_dim0_stride[31:0]
  g1[6] = 0;                                     // stride[47:32]=0 | dim1_stride lo=0
  g1[7] = 0;
  const v4i z4 = {0, 0, 0, 0};
#if TDM_6ARG
  const v8i z8 = {0, 0, 0, 0, 0, 0, 0, 0};
  __builtin_amdgcn_tensor_load_to_lds(g0, g1, z4, z4, z8, 0);
#else
  __builtin_amdgcn_tensor_load_to_lds(g0, g1, z4, z4, 0);
#endif
}
#endif  // HAVE_TDM

// ---------------------------------------------------------------------------
// bf16 WMMA GEMM:  C[M,N] = A[M,K] * B[K,N],  B stored as Bt[N,K] row-major.
// Block 256 thr = 8 waves (2x4); block tile 128x256; wave tile 64x64; BK=32.
// TDM double-buffers A/B tiles into padded LDS; waves read fragments with
// ds_load_b128 and issue 16x v_wmma_f32_16x16x32_bf16 per k-step.
// MODE 0: act=elu(u+b1), ap=elu'   MODE 1: z = c+b2+x
// MODE 2: t = c * ap               MODE 3: w_next=c; sd[m] += coeff*sum_n c*v
// ---------------------------------------------------------------------------
template <int MODE>
__global__ __launch_bounds__(256) void gemm_bf16(
    const bf16_t* __restrict__ A, int lda,
    const bf16_t* __restrict__ Bt, int ldb,
    int M, int N, int K,
    const float* __restrict__ bias,
    const bf16_t* __restrict__ ap,
    const float* __restrict__ faux,
    float* __restrict__ fout,
    bf16_t* __restrict__ bout0,
    bf16_t* __restrict__ bout1,
    float coeff) {
  (void)M;
  __shared__ __align__(64) bf16_t As0[128 * LDSTRIDE];
  __shared__ __align__(64) bf16_t As1[128 * LDSTRIDE];
  __shared__ __align__(64) bf16_t Bs0[256 * LDSTRIDE];
  __shared__ __align__(64) bf16_t Bs1[256 * LDSTRIDE];

  const int lane = threadIdx.x & 31;
  const int wave = threadIdx.x >> 5;
  const int wm = wave >> 2;                 // 0..1
  const int wn = wave & 3;                  // 0..3
  const int m0 = blockIdx.y * 128;
  const int n0 = blockIdx.x * 256;
  const int wm0 = m0 + wm * 64;
  const int wn0 = n0 + wn * 64;
  const int r  = lane & 15;
  const int hi = lane >> 4;

  v8f c[4][4];
  const v8f vzero = {0.f, 0.f, 0.f, 0.f, 0.f, 0.f, 0.f, 0.f};
#pragma unroll
  for (int i = 0; i < 4; ++i)
#pragma unroll
    for (int j = 0; j < 4; ++j) c[i][j] = vzero;

#if HAVE_TDM
  if (threadIdx.x < 32) {
    tdm_load_2d(lds_byte_off(As0), A + (long)m0 * lda, BK, 128, (u32)K, 128u, (u32)lda);
    tdm_load_2d(lds_byte_off(Bs0), Bt + (long)n0 * ldb, BK, 256, (u32)K, 256u, (u32)ldb);
  }
  int buf = 0;
#endif

  for (int k0 = 0; k0 < K; k0 += BK) {
#if HAVE_TDM
    if (threadIdx.x < 32) __builtin_amdgcn_s_wait_tensorcnt(0);
    __syncthreads();
    if ((k0 + BK < K) && threadIdx.x < 32) {
      bf16_t* An = buf ? As0 : As1;
      bf16_t* Bn = buf ? Bs0 : Bs1;
      tdm_load_2d(lds_byte_off(An), A + (long)m0 * lda + (k0 + BK), BK, 128, (u32)K, 128u, (u32)lda);
      tdm_load_2d(lds_byte_off(Bn), Bt + (long)n0 * ldb + (k0 + BK), BK, 256, (u32)K, 256u, (u32)ldb);
    }
    const bf16_t* Ab = buf ? As1 : As0;
    const bf16_t* Bb = buf ? Bs1 : Bs0;
#else
    __syncthreads();
    {  // cooperative copy fallback (single buffer)
      int cch = threadIdx.x;                 // A: 128 rows x 2 halves
      int row = cch >> 1, half = cch & 1;
      const bf16_t* src = A + (long)(m0 + row) * lda + k0 + half * 16;
      bf16_t* dst = As0 + row * LDSTRIDE + half * 16;
      *(v8bf*)dst = *(const v8bf*)src;
      *(v8bf*)(dst + 8) = *(const v8bf*)(src + 8);
#pragma unroll
      for (int it = 0; it < 2; ++it) {       // B: 256 rows x 2 halves
        int cb = threadIdx.x + it * 256;
        int rowb = cb >> 1, halfb = cb & 1;
        const bf16_t* srcb = Bt + (long)(n0 + rowb) * ldb + k0 + halfb * 16;
        bf16_t* dstb = Bs0 + rowb * LDSTRIDE + halfb * 16;
        *(v8bf*)dstb = *(const v8bf*)srcb;
        *(v8bf*)(dstb + 8) = *(const v8bf*)(srcb + 8);
      }
    }
    __syncthreads();
    const bf16_t* Ab = As0;
    const bf16_t* Bb = Bs0;
#endif

    v16bf afr[4], bfr[4];
#pragma unroll
    for (int i = 0; i < 4; ++i) {
      // A 16x32 frag: lane (r,hi) holds K = hi*8+[0..7] and 16+hi*8+[0..7]
      const bf16_t* pa = Ab + (wm * 64 + i * 16 + r) * LDSTRIDE;
      union { v16bf v; v8bf h[2]; } ua;
      ua.h[0] = *(const v8bf*)(pa + hi * 8);
      ua.h[1] = *(const v8bf*)(pa + 16 + hi * 8);
      afr[i] = ua.v;
      // B 32x16 frag: lane holds col, K = hi*8+[0..7] and 16+hi*8+[0..7]
      const bf16_t* pb = Bb + (wn * 64 + i * 16 + r) * LDSTRIDE;
      union { v16bf v; v8bf h[2]; } ub;
      ub.h[0] = *(const v8bf*)(pb + hi * 8);
      ub.h[1] = *(const v8bf*)(pb + 16 + hi * 8);
      bfr[i] = ub.v;
    }
#pragma unroll
    for (int i = 0; i < 4; ++i)
#pragma unroll
      for (int j = 0; j < 4; ++j)
        c[i][j] = __builtin_amdgcn_wmma_f32_16x16x32_bf16(
            false, afr[i], false, bfr[j], (short)0, c[i][j], false, false);
#if HAVE_TDM
    buf ^= 1;
#endif
  }

  // ---- epilogue ----
  float psum[4][8];
  if (MODE == 3) {
#pragma unroll
    for (int i = 0; i < 4; ++i)
#pragma unroll
      for (int rr = 0; rr < 8; ++rr) psum[i][rr] = 0.0f;
  }

#pragma unroll
  for (int mi = 0; mi < 4; ++mi) {
#pragma unroll
    for (int ni = 0; ni < 4; ++ni) {
#pragma unroll
      for (int rr = 0; rr < 8; ++rr) {
        const int m = wm0 + mi * 16 + hi * 8 + rr;
        const int n = wn0 + ni * 16 + r;
        const long idx = (long)m * N + n;
        float val = c[mi][ni][rr];
        if (MODE == 0) {
          val += bias[n];
          const float e = (val > 0.0f) ? val : (__expf(val) - 1.0f);
          bout0[idx] = (bf16_t)e;                                   // elu(u)
          bout1[idx] = (bf16_t)((val > 0.0f) ? 1.0f : (e + 1.0f));  // elu'(u)
        } else if (MODE == 1) {
          val += bias[n] + faux[idx];
          fout[idx] = val;                                          // z = x + g
        } else if (MODE == 2) {
          val *= (float)ap[idx];
          bout0[idx] = (bf16_t)val;                                 // t
        } else {
          bout0[idx] = (bf16_t)val;                                 // w_next
          psum[mi][rr] += val * faux[idx];                          // w_next . v
        }
      }
    }
  }

  if (MODE == 3) {
#pragma unroll
    for (int off = 1; off < 16; off <<= 1)
#pragma unroll
      for (int i = 0; i < 4; ++i)
#pragma unroll
        for (int rr = 0; rr < 8; ++rr)
          psum[i][rr] += __shfl_xor(psum[i][rr], off, 32);
    if (r == 0) {
#pragma unroll
      for (int i = 0; i < 4; ++i)
#pragma unroll
        for (int rr = 0; rr < 8; ++rr)
          atomicAdd(&fout[wm0 + i * 16 + hi * 8 + rr], coeff * psum[i][rr]);
    }
  }
}

// ---------------------------------------------------------------------------
// Host launch
// ---------------------------------------------------------------------------
extern "C" void kernel_launch(void* const* d_in, const int* in_sizes, int n_in,
                              void* d_out, int out_size, void* d_ws, size_t ws_size,
                              hipStream_t stream) {
  (void)n_in; (void)out_size; (void)ws_size;
  const float* x     = (const float*)d_in[0];
  const float* logdf = (const float*)d_in[1];
  const float* W1    = (const float*)d_in[2];
  const float* b1    = (const float*)d_in[3];
  const float* W2    = (const float*)d_in[4];
  const float* b2    = (const float*)d_in[5];
  const float* v     = (const float*)d_in[6];

  const int D  = in_sizes[5];                                // 1024
  const int H  = in_sizes[3];                                // 4096
  const int Bn = in_sizes[1];                                // 8192
  const int S  = (int)((long)in_sizes[6] / ((long)Bn * D));  // 4
  const int NPS = 8;

  float* zout  = (float*)d_out;
  float* ldout = zout + (long)Bn * D;

  uintptr_t wp = (uintptr_t)d_ws;
  auto carve = [&](size_t bytes) -> void* {
    void* q = (void*)wp;
    wp += (bytes + 255) & ~(size_t)255;
    return q;
  };
  bf16_t* xbf   = (bf16_t*)carve((size_t)Bn * D * 2);
  bf16_t* w1bf  = (bf16_t*)carve((size_t)D * H * 2);   // W1 [D,H]   (Bt for t@W1^T)
  bf16_t* w1t   = (bf16_t*)carve((size_t)H * D * 2);   // W1^T [H,D] (Bt for x@W1)
  bf16_t* w2bf  = (bf16_t*)carve((size_t)H * D * 2);   // W2 [H,D]   (Bt for w@W2^T)
  bf16_t* w2t   = (bf16_t*)carve((size_t)D * H * 2);   // W2^T [D,H] (Bt for act@W2)
  bf16_t* actbf = (bf16_t*)carve((size_t)Bn * H * 2);
  bf16_t* apbf  = (bf16_t*)carve((size_t)Bn * H * 2);
  bf16_t* tbf   = (bf16_t*)carve((size_t)Bn * H * 2);
  bf16_t* wa    = (bf16_t*)carve((size_t)Bn * D * 2);
  bf16_t* wb    = (bf16_t*)carve((size_t)Bn * D * 2);
  float*  sd    = (float*)carve((size_t)Bn * 4);

  const dim3 blk(256);

  k_zero_f32<<<(Bn + 255) / 256, blk, 0, stream>>>(sd, Bn);
  k_cvt_bf16<<<(unsigned)(((long)Bn * D + 255) / 256), blk, 0, stream>>>(x, xbf, (long)Bn * D);
  k_cvt_bf16<<<(unsigned)(((long)D * H + 255) / 256), blk, 0, stream>>>(W1, w1bf, (long)D * H);
  k_cvt_bf16<<<(unsigned)(((long)H * D + 255) / 256), blk, 0, stream>>>(W2, w2bf, (long)H * D);
  k_cvt_tr_bf16<<<(unsigned)(((long)D * H + 255) / 256), blk, 0, stream>>>(W1, w1t, D, H);
  k_cvt_tr_bf16<<<(unsigned)(((long)H * D + 255) / 256), blk, 0, stream>>>(W2, w2t, H, D);

  // forward: u = x@W1 (+b1) -> act, elu'
  gemm_bf16<0><<<dim3(H / 256, Bn / 128), blk, 0, stream>>>(
      xbf, D, w1t, D, Bn, H, D, b1, nullptr, nullptr, nullptr, actbf, apbf, 0.0f);
  // z = act@W2 + b2 + x
  gemm_bf16<1><<<dim3(D / 256, Bn / 128), blk, 0, stream>>>(
      actbf, H, w2t, H, Bn, D, H, b2, nullptr, x, zout, nullptr, nullptr, 0.0f);

  // power series per probe
  for (int s = 0; s < S; ++s) {
    const float* vs = v + (long)s * Bn * D;
    k_cvt_bf16<<<(unsigned)(((long)Bn * D + 255) / 256), blk, 0, stream>>>(vs, wa, (long)Bn * D);
    bf16_t* cur = wa;
    bf16_t* nxt = wb;
    for (int k = 1; k <= NPS; ++k) {
      // t = (w @ W2^T) .* elu'(u)
      gemm_bf16<2><<<dim3(H / 256, Bn / 128), blk, 0, stream>>>(
          cur, D, w2bf, D, Bn, H, D, nullptr, apbf, nullptr, nullptr, tbf, nullptr, 0.0f);
      // w_next = t @ W1^T ; sumdiag += sign/k/S * <w_next, v>
      const float coeff = (((k & 1) ? 1.0f : -1.0f) / (float)k) / (float)S;
      gemm_bf16<3><<<dim3(D / 256, Bn / 128), blk, 0, stream>>>(
          tbf, H, w1bf, H, Bn, D, H, nullptr, nullptr, vs, sd, nxt, nullptr, coeff);
      bf16_t* tmp = cur; cur = nxt; nxt = tmp;
    }
  }

  k_final<<<(Bn + 255) / 256, blk, 0, stream>>>(logdf, sd, ldout, Bn);
}
